// GCN_81887846465661
// MI455X (gfx1250) — matrix operand order repaired and checked by reading, not compiled, not run on hardware
//
#include <hip/hip_runtime.h>
#include <hip/hip_bf16.h>

// ---- problem constants (match reference) ----
#define NN 100000   // nodes (== 16 * 6250)
#define EE 1000000  // edges
#define PP 100000   // pos/neg pairs (== 16 * 6250)
#define KIN 128
#define HH  64

typedef __attribute__((ext_vector_type(16))) _Float16 v16h;
typedef __attribute__((ext_vector_type(8)))  float    v8f;

// ---------------- degree / norm ----------------
__global__ __launch_bounds__(256)
void deg_kernel(const int* __restrict__ src, const int* __restrict__ dst,
                float* __restrict__ deg_out, float* __restrict__ deg_in) {
    int e = blockIdx.x * blockDim.x + threadIdx.x;
    if (e < EE) {
        atomicAdd(&deg_out[src[e]], 1.0f);
        atomicAdd(&deg_in[dst[e]], 1.0f);
    }
}

__global__ __launch_bounds__(256)
void norm_kernel(float* __restrict__ a, float* __restrict__ b) {
    int i = blockIdx.x * blockDim.x + threadIdx.x;
    if (i < NN) {
        a[i] = rsqrtf(fmaxf(a[i], 1.0f));
        b[i] = rsqrtf(fmaxf(b[i], 1.0f));
    }
}

// ---------------- pack K x 64 fp32 weights into WMMA B-operand lane layout ----------------
// For (kc, nt, lane): 16 contiguous f16 = W[kc*32 + (lane>=16?8:0) + (j&7) + (j>=8?16:0)][nt*16 + (lane&15)]
// so each lane's v16h B operand becomes ONE contiguous 32-byte load.
__global__ __launch_bounds__(256)
void pack_b_kernel(const float* __restrict__ W, _Float16* __restrict__ Wp, int K) {
    int idx = blockIdx.x * blockDim.x + threadIdx.x;   // K*64 elements
    if (idx >= K * HH) return;
    const int j    = idx & 15;
    const int lane = (idx >> 4) & 31;
    const int t    = idx >> 9;
    const int nt   = t & 3;
    const int kc   = t >> 2;
    const int k    = kc * 32 + ((lane >> 4) ? 8 : 0) + (j & 7) + ((j >> 3) ? 16 : 0);
    const int n    = nt * 16 + (lane & 15);
    Wp[idx] = (_Float16)W[k * HH + n];
}

// ---------------- hw = (h * norm_src) @ W  via WMMA f16 ----------------
// One wave per 16-row tile; B operands pre-packed and PRELOADED into registers
// (loop-invariant), so the k-loop only waits on activation loads.
template <int K>
__global__ __launch_bounds__(256)
void gcn_gemm(const float* __restrict__ h_in, const _Float16* __restrict__ Wp,
              const float* __restrict__ norm_src, float* __restrict__ hw) {
    const int wave = threadIdx.x >> 5;
    const int lane = threadIdx.x & 31;
    const int tile = blockIdx.x * 8 + wave;
    if (tile >= NN / 16) return;

    const int row0 = tile * 16;
    const int row  = row0 + (lane & 15);
    const int col  = lane & 15;
    const int hiA  = (lane >> 4) ? 8 : 0;
    const float* rp = h_in + (size_t)row * K;
    const float nrm = norm_src[row];
    const v16h* bp = (const v16h*)Wp;   // [(kc*4+nt)*32 + lane]

    // preload all weight B-operands (K=64: 8 v16h, K=128: 16 v16h)
    v16h breg[(K / 32) * 4];
#pragma unroll
    for (int t = 0; t < (K / 32) * 4; ++t)
        breg[t] = bp[t * 32 + lane];

    const v8f zero = {0.f,0.f,0.f,0.f,0.f,0.f,0.f,0.f};
    v8f acc[4] = {zero, zero, zero, zero};

#pragma unroll
    for (int kc = 0; kc < K / 32; ++kc) {
        v16h a;
        const int kb = kc * 32 + hiA;
#pragma unroll
        for (int j = 0; j < 8; ++j) a[j]     = (_Float16)(rp[kb + j]      * nrm);
#pragma unroll
        for (int j = 0; j < 8; ++j) a[j + 8] = (_Float16)(rp[kb + 16 + j] * nrm);

#pragma unroll
        for (int nt = 0; nt < 4; ++nt)
            acc[nt] = __builtin_amdgcn_wmma_f32_16x16x32_f16(
                false, a, false, breg[kc * 4 + nt], (short)0, acc[nt], false, false);
    }

    const int mOff = (lane >> 4) * 8;
#pragma unroll
    for (int nt = 0; nt < 4; ++nt)
#pragma unroll
        for (int r = 0; r < 8; ++r)
            hw[(size_t)(row0 + mOff + r) * HH + nt * 16 + col] = acc[nt][r];
}

// ---------------- edge pass: agg[dst] += hw[src] * ew ----------------
// 64 consecutive threads per edge: coalesced row read + coalesced atomics (L2-resident).
__global__ __launch_bounds__(256)
void edge_kernel(const float* __restrict__ hw, const float* __restrict__ ew,
                 const int* __restrict__ src, const int* __restrict__ dst,
                 float* __restrict__ agg) {
    long long idx = (long long)blockIdx.x * blockDim.x + threadIdx.x;
    if (idx >= (long long)EE * HH) return;
    const int e = (int)(idx >> 6);
    const int f = (int)(idx & 63);
    const float v = hw[(size_t)src[e] * HH + f] * ew[e];
    atomicAdd(&agg[(size_t)dst[e] * HH + f], v);
}

// ---------------- finalize: h = agg * norm_dst + b (optional relu) ----------------
__global__ __launch_bounds__(256)
void fin_kernel(const float* __restrict__ agg, const float* __restrict__ norm_dst,
                const float* __restrict__ bias, float* __restrict__ h_out, int do_relu) {
    long long idx = (long long)blockIdx.x * blockDim.x + threadIdx.x;
    if (idx >= (long long)NN * HH) return;
    const int i = (int)(idx >> 6);
    const int f = (int)(idx & 63);
    float v = agg[idx] * norm_dst[i] + bias[f];
    if (do_relu) v = fmaxf(v, 0.0f);
    h_out[idx] = v;
}

// ---------------- MLP on node-pair products: two WMMA GEMMs + dot ----------------
// 8 waves/block, one 16-pair tile per wave; packed+preloaded P0/P1 B operands;
// per-wave LDS staging tile for the C-layout -> A-layout relayout between GEMMs.
__global__ __launch_bounds__(256)
void mlp_kernel(const float* __restrict__ h,
                const int* __restrict__ sidx, const int* __restrict__ didx,
                const _Float16* __restrict__ Pp0, const float* __restrict__ pb0,
                const _Float16* __restrict__ Pp1, const float* __restrict__ pb1,
                const float* __restrict__ P2, const float* __restrict__ pb2,
                float* __restrict__ out) {
    __shared__ _Float16 zs[8][16 * HH];          // 2KB per wave
    const int wave = threadIdx.x >> 5;
    const int lane = threadIdx.x & 31;
    const int tile = blockIdx.x * 8 + wave;
    const bool active = tile < PP / 16;

    const int row  = tile * 16 + (lane & 15);
    const int col  = lane & 15;
    const int hiA  = (lane >> 4) ? 8 : 0;
    const int mOff = (lane >> 4) * 8;

    const v8f zero = {0.f,0.f,0.f,0.f,0.f,0.f,0.f,0.f};
    v8f acc[4] = {zero, zero, zero, zero};

    // preload both weight operand sets (8 + 8 v16h) so GEMM2 weights are in
    // flight during GEMM1 and the LDS relayout
    v16h b0r[8], b1r[8];
#pragma unroll
    for (int t = 0; t < 8; ++t) b0r[t] = ((const v16h*)Pp0)[t * 32 + lane];
#pragma unroll
    for (int t = 0; t < 8; ++t) b1r[t] = ((const v16h*)Pp1)[t * 32 + lane];

    if (active) {
        const float* hs = h + (size_t)sidx[row] * HH;
        const float* hd = h + (size_t)didx[row] * HH;

        // GEMM 1: z @ P0  (z = hs*hd computed on the fly)
#pragma unroll
        for (int kc = 0; kc < 2; ++kc) {
            v16h a;
            const int kb = kc * 32 + hiA;
#pragma unroll
            for (int j = 0; j < 8; ++j) a[j]     = (_Float16)(hs[kb + j]      * hd[kb + j]);
#pragma unroll
            for (int j = 0; j < 8; ++j) a[j + 8] = (_Float16)(hs[kb + 16 + j] * hd[kb + 16 + j]);
#pragma unroll
            for (int nt = 0; nt < 4; ++nt)
                acc[nt] = __builtin_amdgcn_wmma_f32_16x16x32_f16(
                    false, a, false, b0r[kc * 4 + nt], (short)0, acc[nt], false, false);
        }

        // bias + relu, relayout through LDS (C layout -> row-major)
#pragma unroll
        for (int nt = 0; nt < 4; ++nt) {
            const float bb = pb0[nt * 16 + col];
#pragma unroll
            for (int r = 0; r < 8; ++r)
                zs[wave][(mOff + r) * HH + nt * 16 + col] =
                    (_Float16)fmaxf(acc[nt][r] + bb, 0.0f);
        }
    }
    __syncthreads();                              // uniform across block
    if (!active) return;

    // GEMM 2: z1 @ P1
    v8f acc2[4] = {zero, zero, zero, zero};
#pragma unroll
    for (int kc = 0; kc < 2; ++kc) {
        v16h a;
        const int kb = kc * 32 + hiA;
        const _Float16* zr = &zs[wave][(lane & 15) * HH];
#pragma unroll
        for (int j = 0; j < 8; ++j) a[j]     = zr[kb + j];
#pragma unroll
        for (int j = 0; j < 8; ++j) a[j + 8] = zr[kb + 16 + j];
#pragma unroll
        for (int nt = 0; nt < 4; ++nt)
            acc2[nt] = __builtin_amdgcn_wmma_f32_16x16x32_f16(
                false, a, false, b1r[kc * 4 + nt], (short)0, acc2[nt], false, false);
    }

    // bias + relu + dot with P2, reduce across the 16-lane half-groups
    float partial[8];
#pragma unroll
    for (int r = 0; r < 8; ++r) partial[r] = 0.0f;
#pragma unroll
    for (int nt = 0; nt < 4; ++nt) {
        const float p2 = P2[nt * 16 + col];
        const float bb = pb1[nt * 16 + col];
#pragma unroll
        for (int r = 0; r < 8; ++r)
            partial[r] += fmaxf(acc2[nt][r] + bb, 0.0f) * p2;
    }
#pragma unroll
    for (int m = 1; m < 16; m <<= 1)
#pragma unroll
        for (int r = 0; r < 8; ++r)
            partial[r] += __shfl_xor(partial[r], m, 32);

    if (col == 0) {
        const float c = pb2[0];
#pragma unroll
        for (int r = 0; r < 8; ++r)
            out[tile * 16 + mOff + r] = partial[r] + c;
    }
}

// ---------------- host side ----------------
extern "C" void kernel_launch(void* const* d_in, const int* in_sizes, int n_in,
                              void* d_out, int out_size, void* d_ws, size_t ws_size,
                              hipStream_t stream) {
    (void)in_sizes; (void)n_in; (void)out_size; (void)ws_size;

    const float* x       = (const float*)d_in[0];
    const float* ew      = (const float*)d_in[1];   // [3, E]
    const int*   src     = (const int*)d_in[2];
    const int*   dst     = (const int*)d_in[3];
    const int*   pos_src = (const int*)d_in[4];
    const int*   pos_dst = (const int*)d_in[5];
    const int*   neg_src = (const int*)d_in[6];
    const int*   neg_dst = (const int*)d_in[7];
    const float* W0 = (const float*)d_in[8];
    const float* b0 = (const float*)d_in[9];
    const float* W1 = (const float*)d_in[10];
    const float* b1 = (const float*)d_in[11];
    const float* W2 = (const float*)d_in[12];
    const float* b2 = (const float*)d_in[13];
    const float* P0 = (const float*)d_in[14];
    const float* pb0 = (const float*)d_in[15];
    const float* P1 = (const float*)d_in[16];
    const float* pb1 = (const float*)d_in[17];
    const float* P2 = (const float*)d_in[18];
    const float* pb2 = (const float*)d_in[19];
    float* out = (float*)d_out;

    // workspace (floats): norms(2N) | hw(64N) | agg(64N) | h(64N) | packed weights
    float* ws       = (float*)d_ws;
    float* norm_src = ws;
    float* norm_dst = ws + NN;
    float* hw   = ws + 2 * (size_t)NN;
    float* agg  = hw + (size_t)NN * HH;
    float* hbuf = agg + (size_t)NN * HH;
    _Float16* Wp0 = (_Float16*)(hbuf + (size_t)NN * HH);   // 128*64 f16
    _Float16* Wp1 = Wp0 + (size_t)KIN * HH;                // 64*64 f16
    _Float16* Wp2 = Wp1 + (size_t)HH * HH;
    _Float16* Pp0 = Wp2 + (size_t)HH * HH;
    _Float16* Pp1 = Pp0 + (size_t)HH * HH;

    const int TILES = NN / 16;                 // 6250 (exact; PP/16 identical)
    const int WGRID = (TILES + 7) / 8;
    const int EDGE_GRID = (int)(((long long)EE * HH + 255) / 256);
    const int NH_GRID   = (int)(((long long)NN * HH + 255) / 256);

    // pack all weight matrices into WMMA B-operand layout (cheap, every call)
    pack_b_kernel<<<(KIN * HH + 255) / 256, 256, 0, stream>>>(W0, Wp0, KIN);
    pack_b_kernel<<<(HH * HH + 255) / 256, 256, 0, stream>>>(W1, Wp1, HH);
    pack_b_kernel<<<(HH * HH + 255) / 256, 256, 0, stream>>>(W2, Wp2, HH);
    pack_b_kernel<<<(HH * HH + 255) / 256, 256, 0, stream>>>(P0, Pp0, HH);
    pack_b_kernel<<<(HH * HH + 255) / 256, 256, 0, stream>>>(P1, Pp1, HH);

    // degrees -> norms (zeroed every call for determinism)
    hipMemsetAsync(norm_src, 0, 2 * (size_t)NN * sizeof(float), stream);
    deg_kernel<<<(EE + 255) / 256, 256, 0, stream>>>(src, dst, norm_src, norm_dst);
    norm_kernel<<<(NN + 255) / 256, 256, 0, stream>>>(norm_src, norm_dst);

    // ---- layer 0 (K=128, relu) ----
    gcn_gemm<KIN><<<WGRID, 256, 0, stream>>>(x, Wp0, norm_src, hw);
    hipMemsetAsync(agg, 0, (size_t)NN * HH * sizeof(float), stream);
    edge_kernel<<<EDGE_GRID, 256, 0, stream>>>(hw, ew + 0 * (size_t)EE, src, dst, agg);
    fin_kernel<<<NH_GRID, 256, 0, stream>>>(agg, norm_dst, b0, hbuf, 1);

    // ---- layer 1 (K=64, relu) ----
    gcn_gemm<HH><<<WGRID, 256, 0, stream>>>(hbuf, Wp1, norm_src, hw);
    hipMemsetAsync(agg, 0, (size_t)NN * HH * sizeof(float), stream);
    edge_kernel<<<EDGE_GRID, 256, 0, stream>>>(hw, ew + 1 * (size_t)EE, src, dst, agg);
    fin_kernel<<<NH_GRID, 256, 0, stream>>>(agg, norm_dst, b1, hbuf, 1);

    // ---- layer 2 (K=64, no relu) ----
    gcn_gemm<HH><<<WGRID, 256, 0, stream>>>(hbuf, Wp2, norm_src, hw);
    hipMemsetAsync(agg, 0, (size_t)NN * HH * sizeof(float), stream);
    edge_kernel<<<EDGE_GRID, 256, 0, stream>>>(hw, ew + 2 * (size_t)EE, src, dst, agg);
    fin_kernel<<<NH_GRID, 256, 0, stream>>>(agg, norm_dst, b2, hbuf, 0);

    // ---- link predictor MLP: pos then neg ----
    mlp_kernel<<<WGRID, 256, 0, stream>>>(hbuf, pos_src, pos_dst,
                                          Pp0, pb0, Pp1, pb1, P2, pb2, out);
    mlp_kernel<<<WGRID, 256, 0, stream>>>(hbuf, neg_src, neg_dst,
                                          Pp0, pb0, Pp1, pb1, P2, pb2, out + PP);
}
